// CharLSTM_13752485281997
// MI455X (gfx1250) — matrix operand order repaired and checked by reading, not compiled, not run on hardware
//
#include <hip/hip_runtime.h>
#include <hip/hip_bf16.h>

// ---------------------------------------------------------------------------
// CharLSTM on gfx1250 (MI455X, wave32, WMMA).
//   Phase 1: weights + emb_table fp32 -> bf16 (all weights ~44MB: L2-resident)
//   Phase 2: input GEMM  X[g][t][:] = embs @ Wg[:, :E]^T + bg  via v_wmma bf16
//   Phase 3: persistent cooperative kernel, 4096 sequential steps:
//            h staged to LDS each step, gate matvec via WMMA with h broadcast
//            into all 16 B columns, two-bank software-pipelined operand loads
//            (partial s_wait_loadcnt instead of full drains), 4 accumulator
//            chains, X[t] prefetched ahead of the barrier, Wy@h via WMMA.
// ---------------------------------------------------------------------------

static constexpr int SEQ = 4096;
static constexpr int EMB = 512;
static constexpr int HID = 2048;
static constexpr int NC  = 256;
static constexpr int RS  = EMB + HID;   // 2560 row stride of Wf/Wi/Wo/Wc

static constexpr int NWG = 64;          // persistent kernel workgroups
static constexpr int BLK = 256;         // 8 waves per workgroup -> 512 waves

// ---- workspace layout (bytes) ----
static constexpr size_t WS_WG  = 0;                                   // 4*HID*RS bf16
static constexpr size_t WS_WY  = WS_WG  + (size_t)4 * HID * RS * 2;   // NC*HID bf16
static constexpr size_t WS_EMB = WS_WY  + (size_t)NC * HID * 2;       // NC*EMB bf16
static constexpr size_t WS_X   = WS_EMB + (size_t)NC * EMB * 2;       // 4*SEQ*HID bf16
static constexpr size_t WS_HBF = WS_X   + (size_t)4 * SEQ * HID * 2;  // HID bf16
static constexpr size_t WS_HF  = WS_HBF + (size_t)HID * 2;            // HID f32
static constexpr size_t WS_CF  = WS_HF  + (size_t)HID * 4;            // HID f32
static constexpr size_t WS_G   = WS_CF  + (size_t)HID * 4;            // 4*HID f32 gate matvec
static constexpr size_t WS_BAR = WS_G   + (size_t)4 * HID * 4;        // 2 u32 barrier
static constexpr size_t WS_END = WS_BAR + 64;

typedef __attribute__((ext_vector_type(16))) __bf16 v16bf;
typedef __attribute__((ext_vector_type(8)))  float  v8f;

union BF16x16 {
    v16bf v;
    unsigned short u[16];
    uint4 q[2];
};

__device__ __forceinline__ unsigned short f2bf(float f) {
    unsigned u = __float_as_uint(f);
    u += 0x7FFFu + ((u >> 16) & 1u);          // round-to-nearest-even
    return (unsigned short)(u >> 16);
}
__device__ __forceinline__ float bf2f(unsigned short s) {
    return __uint_as_float(((unsigned)s) << 16);
}
__device__ __forceinline__ float sigmoidf(float x) {
    return 1.0f / (1.0f + __expf(-x));
}

// ---------------------------------------------------------------------------
// fp32 -> bf16 conversion: Wf,Wi,Wo,Wc concatenated; Wy; emb_table
// ---------------------------------------------------------------------------
__global__ void convert_inputs(const float* __restrict__ Wf, const float* __restrict__ Wi,
                               const float* __restrict__ Wo, const float* __restrict__ Wc,
                               const float* __restrict__ Wy, const float* __restrict__ Em,
                               unsigned short* __restrict__ Wg_bf,
                               unsigned short* __restrict__ Wy_bf,
                               unsigned short* __restrict__ Em_bf) {
    const size_t n1 = (size_t)4 * HID * RS;
    const size_t n2 = (size_t)NC * HID;
    const size_t n3 = (size_t)NC * EMB;
    const size_t stride = (size_t)gridDim.x * blockDim.x;
    for (size_t i = (size_t)blockIdx.x * blockDim.x + threadIdx.x; i < n1 + n2 + n3;
         i += stride) {
        if (i < n1) {
            int g = (int)(i / ((size_t)HID * RS));
            size_t r = i % ((size_t)HID * RS);
            const float* src = (g == 0) ? Wf : (g == 1) ? Wi : (g == 2) ? Wo : Wc;
            Wg_bf[i] = f2bf(src[r]);
        } else if (i < n1 + n2) {
            size_t r = i - n1;
            Wy_bf[r] = f2bf(Wy[r]);
        } else {
            size_t r = i - n1 - n2;
            Em_bf[r] = f2bf(Em[r]);
        }
    }
}

// ---------------------------------------------------------------------------
// init h, c, barrier state
// ---------------------------------------------------------------------------
__global__ void init_state(unsigned short* __restrict__ hbf, float* __restrict__ hf,
                           float* __restrict__ cf, unsigned* __restrict__ bar) {
    for (int i = threadIdx.x; i < HID; i += blockDim.x) {
        hbf[i] = 0; hf[i] = 0.0f; cf[i] = 0.0f;
    }
    if (threadIdx.x < 2) bar[threadIdx.x] = 0u;
}

// ---------------------------------------------------------------------------
// Operand loads for one 16x16x32 bf16 WMMA K-chunk.
//   A pattern: {kk+aoff, kk+aoff+16} 16B each  (aoff = hi-lane ? 8 : 0)
//   B pattern: {kk+boff/2, kk+boff/2+16}  16B each  (boff = hi-lane ? 16 : 0)
// ---------------------------------------------------------------------------
__device__ __forceinline__ void load_a(BF16x16& a, const unsigned short* p,
                                       int kk, int aoff) {
    a.q[0] = *(const uint4*)(p + kk + aoff);
    a.q[1] = *(const uint4*)(p + kk + aoff + 16);
}
__device__ __forceinline__ void load_b(BF16x16& b, const unsigned short* p,
                                       int kk, int boff) {
    b.q[0] = *(const uint4*)(p + kk + boff / 2);
    b.q[1] = *(const uint4*)(p + kk + boff / 2 + 16);
}

// ---------------------------------------------------------------------------
// Two-bank software-pipelined dot: while bank X's 4 WMMAs execute, bank Y's
// 16 loads are in flight (partial loadcnt waits, one 128-K block prefetch
// distance). 4 independent accumulator chains. K % 256 == 0.
// ---------------------------------------------------------------------------
__device__ __forceinline__ v8f wmma_dot_pipe(const unsigned short* aptr,
                                             const unsigned short* bptr,
                                             int K, int aoff, int boff) {
    BF16x16 A0[4], B0[4], A1[4], B1[4];
    v8f acc[4] = {};
#pragma unroll
    for (int i = 0; i < 4; ++i) {
        load_a(A0[i], aptr, i * 32, aoff);
        load_b(B0[i], bptr, i * 32, boff);
    }
    for (int kc = 0; kc < K; kc += 256) {
#pragma unroll
        for (int i = 0; i < 4; ++i) {          // prefetch bank1 (kc+128)
            load_a(A1[i], aptr, kc + 128 + i * 32, aoff);
            load_b(B1[i], bptr, kc + 128 + i * 32, boff);
        }
#pragma unroll
        for (int i = 0; i < 4; ++i)            // compute bank0 (kc)
            acc[i] = __builtin_amdgcn_wmma_f32_16x16x32_bf16(
                false, A0[i].v, false, B0[i].v, (short)0, acc[i], false, false);
        const int kn = (kc + 256 < K) ? (kc + 256) : kc;   // clamp: safe reload
#pragma unroll
        for (int i = 0; i < 4; ++i) {          // prefetch bank0 (kc+256)
            load_a(A0[i], aptr, kn + i * 32, aoff);
            load_b(B0[i], bptr, kn + i * 32, boff);
        }
#pragma unroll
        for (int i = 0; i < 4; ++i)            // compute bank1 (kc+128)
            acc[i] = __builtin_amdgcn_wmma_f32_16x16x32_bf16(
                false, A1[i].v, false, B1[i].v, (short)0, acc[i], false, false);
    }
    return (acc[0] + acc[1]) + (acc[2] + acc[3]);
}

// ---------------------------------------------------------------------------
// Matvec: 16 rows (stride rs) dot h (broadcast into all 16 B columns).
// lane 0 holds rows m0..m0+7, lane 16 rows m0+8..+15 (all columns identical).
// ---------------------------------------------------------------------------
__device__ __forceinline__ v8f wmma_matvec16(const unsigned short* __restrict__ w0,
                                             size_t rs, int K,
                                             const unsigned short* hsrc,  // LDS
                                             int lane) {
    const unsigned short* wrow = w0 + (size_t)(lane & 15) * rs;
    return wmma_dot_pipe(wrow, hsrc, K, (lane >= 16) ? 8 : 0, (lane >= 16) ? 16 : 0);
}

// ---------------------------------------------------------------------------
// Input GEMM: X[g][t][j] = sum_k emb_bf[seq[t]][k] * Wg[j][k] + bg[j]
// One wave per 16x16 tile; K = EMB = 512; pipelined WMMA dot.
// ---------------------------------------------------------------------------
__global__ __launch_bounds__(256) void input_gemm(
        const int* __restrict__ seq, const unsigned short* __restrict__ emb_bf,
        const unsigned short* __restrict__ Wg_bf,
        const float* __restrict__ bfp, const float* __restrict__ bip,
        const float* __restrict__ bop, const float* __restrict__ bcp,
        unsigned short* __restrict__ X) {
    const int lane = threadIdx.x & 31;
    const int tile = (int)((blockIdx.x * (unsigned)blockDim.x + threadIdx.x) >> 5);
    const int MT = SEQ / 16, NT = HID / 16;          // 256, 128
    const int g  = tile / (MT * NT);
    const int rm = tile % (MT * NT);
    const int mt = rm / NT;
    const int nt = rm % NT;

    const int mrow = mt * 16 + (lane & 15);
    const int idx  = seq[mrow];
    const unsigned short* arow = emb_bf + (size_t)idx * EMB;   // A: gathered emb row

    const int nrow = nt * 16 + (lane & 15);
    const unsigned short* wrow = Wg_bf + ((size_t)g * HID + nrow) * RS;   // k in [0,EMB)

    const int aoff = (lane >= 16) ? 8 : 0;
    const int boff = (lane >= 16) ? 16 : 0;

    // A operand = emb row (A layout), B operand = weight row (B layout).
    v8f acc = wmma_dot_pipe(arow, wrow, EMB, aoff, boff);

    const float* bg = (g == 0) ? bfp : (g == 1) ? bip : (g == 2) ? bop : bcp;
    const int ncol = nt * 16 + (lane & 15);
    const float bb = bg[ncol];
#pragma unroll
    for (int r = 0; r < 8; ++r) {
        int m = mt * 16 + r + ((lane >= 16) ? 8 : 0);
        X[((size_t)g * SEQ + m) * HID + ncol] = f2bf(acc[r] + bb);
    }
}

// ---------------------------------------------------------------------------
// device-wide sense-reversing barrier
// ---------------------------------------------------------------------------
__device__ __forceinline__ void grid_barrier(unsigned* bar) {
    __threadfence();
    __syncthreads();
    if (threadIdx.x == 0) {
        volatile unsigned* vcnt = bar;
        volatile unsigned* vgen = bar + 1;
        unsigned g = *vgen;
        unsigned prev = atomicAdd(bar, 1u);
        if (prev == (unsigned)(NWG - 1)) {
            *vcnt = 0u;
            __threadfence();
            *vgen = g + 1u;
        } else {
            while (*vgen == g) { __builtin_amdgcn_s_sleep(1); }
        }
        __threadfence();
    }
    __syncthreads();
}

// ---------------------------------------------------------------------------
// Persistent recurrence kernel: 64 WGs x 256 threads = 512 waves.
// ---------------------------------------------------------------------------
__global__ __launch_bounds__(BLK, 1) void lstm_recurrence(
        const unsigned short* __restrict__ Wg_bf,   // [4*HID][RS], hidden part at +EMB
        const unsigned short* __restrict__ Wy_bf,   // [NC][HID]
        const unsigned short* __restrict__ X,       // [4][SEQ][HID] bf16 pre-activations
        const float* __restrict__ by,
        unsigned short* __restrict__ hbf, float* __restrict__ hf,
        float* __restrict__ cf, float* __restrict__ gates,
        unsigned* __restrict__ bar, float* __restrict__ out) {
    __shared__ __align__(16) unsigned short hlds[HID];   // 4KB staged h (bf16)

    const int lane = threadIdx.x & 31;
    const int gtid = (int)(blockIdx.x * (unsigned)blockDim.x + threadIdx.x);
    const int wave = gtid >> 5;                     // 0..511

    // stage h0 (zeros) into LDS: 256 threads x 16B
    *(uint4*)(hlds + threadIdx.x * 8) = *(const uint4*)(hbf + threadIdx.x * 8);
    __syncthreads();

    for (int t = 0; t < SEQ; ++t) {
        // ---- prefetch X[t] (independent of h/gates): overlaps with region A
        float xf_ = 0.f, xi_ = 0.f, xo_ = 0.f, xc_ = 0.f;
        if (gtid < HID) {
            const size_t xb = (size_t)t * HID + gtid;
            xf_ = bf2f(X[xb]);
            xi_ = bf2f(X[(size_t)1 * SEQ * HID + xb]);
            xo_ = bf2f(X[(size_t)2 * SEQ * HID + xb]);
            xc_ = bf2f(X[(size_t)3 * SEQ * HID + xb]);
        }

        // ---- A: gate matvec [Wfh;Wih;Woh;Wch] @ h_{t-1}, one wave per 16 rows
        {
            const int r0 = wave * 16;               // 0..8191
            const int g  = r0 >> 11;
            const int j0 = r0 & (HID - 1);
            const unsigned short* w0 = Wg_bf + ((size_t)g * HID + j0) * RS + EMB;
            v8f acc = wmma_matvec16(w0, RS, HID, hlds, lane);
            if (lane == 0 || lane == 16) {
                int mb = r0 + ((lane >= 16) ? 8 : 0);
#pragma unroll
                for (int r = 0; r < 8; ++r) gates[mb + r] = acc[r];
            }
        }
        grid_barrier(bar);

        // ---- B: elementwise cell update (2048 lanes across the grid)
        if (gtid < HID) {
            const int j = gtid;
            float ft  = sigmoidf(xf_ + gates[0 * HID + j]);
            float it  = sigmoidf(xi_ + gates[1 * HID + j]);
            float ot  = sigmoidf(xo_ + gates[2 * HID + j]);
            float tc  = tanhf(xc_ + gates[3 * HID + j]);
            float cn  = ft * cf[j] + it * tc;
            float hn  = ot * tanhf(cn);
            cf[j] = cn;
            hf[j] = hn;
            hbf[j] = f2bf(hn);
        }
        grid_barrier(bar);

        // ---- re-stage h_t into LDS for region C and next step's region A
        *(uint4*)(hlds + threadIdx.x * 8) = *(const uint4*)(hbf + threadIdx.x * 8);
        __syncthreads();

        // ---- C: y_t = Wy @ h_t + by (16 waves, 16 rows each)
        if (wave < NC / 16) {
            const int r0 = wave * 16;
            v8f acc = wmma_matvec16(Wy_bf + (size_t)r0 * HID, HID, HID, hlds, lane);
            if (lane == 0 || lane == 16) {
                int mb = r0 + ((lane >= 16) ? 8 : 0);
#pragma unroll
                for (int r = 0; r < 8; ++r)
                    out[(size_t)t * NC + mb + r] = acc[r] + by[mb + r];
            }
        }
        // next step's gates write is ordered behind this step's gate reads by
        // the barrier above; hbf/hlds are written before their readers run.
    }

    // ---- tail: final h, c
    if (gtid < HID) {
        out[(size_t)SEQ * NC + gtid]       = hf[gtid];
        out[(size_t)SEQ * NC + HID + gtid] = cf[gtid];
    }
}

// ---------------------------------------------------------------------------
extern "C" void kernel_launch(void* const* d_in, const int* in_sizes, int n_in,
                              void* d_out, int out_size, void* d_ws, size_t ws_size,
                              hipStream_t stream) {
    const int*   seq = (const int*)  d_in[0];
    const float* emb = (const float*)d_in[1];
    const float* Wf  = (const float*)d_in[2];
    const float* bf_ = (const float*)d_in[3];
    const float* Wi  = (const float*)d_in[4];
    const float* bi_ = (const float*)d_in[5];
    const float* Wo  = (const float*)d_in[6];
    const float* bo_ = (const float*)d_in[7];
    const float* Wc  = (const float*)d_in[8];
    const float* bc_ = (const float*)d_in[9];
    const float* Wy  = (const float*)d_in[10];
    const float* by_ = (const float*)d_in[11];
    float* out = (float*)d_out;

    char* ws = (char*)d_ws;
    unsigned short* Wg_bf = (unsigned short*)(ws + WS_WG);
    unsigned short* Wy_bf = (unsigned short*)(ws + WS_WY);
    unsigned short* Em_bf = (unsigned short*)(ws + WS_EMB);
    unsigned short* X     = (unsigned short*)(ws + WS_X);
    unsigned short* hbf   = (unsigned short*)(ws + WS_HBF);
    float*          hf    = (float*)(ws + WS_HF);
    float*          cf    = (float*)(ws + WS_CF);
    float*          gates = (float*)(ws + WS_G);
    unsigned*       bar   = (unsigned*)(ws + WS_BAR);

    // 1) weights + embedding table -> bf16
    convert_inputs<<<4096, 256, 0, stream>>>(Wf, Wi, Wo, Wc, Wy, emb,
                                             Wg_bf, Wy_bf, Em_bf);
    // 2) state + barrier init (every call: deterministic)
    init_state<<<1, 256, 0, stream>>>(hbf, hf, cf, bar);
    // 3) batched input GEMM: (SEQ/16)*(HID/16)*4 tiles, 1 wave each, 8 waves/block
    input_gemm<<<(SEQ / 16) * (HID / 16) * 4 / 8, 256, 0, stream>>>(
        seq, Em_bf, Wg_bf, bf_, bi_, bo_, bc_, X);
    // 4) persistent sequential recurrence
    lstm_recurrence<<<NWG, BLK, 0, stream>>>(Wg_bf, Wy_bf, X, by_, hbf, hf, cf,
                                             gates, bar, out);
    (void)in_sizes; (void)n_in; (void)out_size; (void)ws_size; (void)WS_END;
}
